// Model_11759620456640
// MI455X (gfx1250) — compile-verified, matrix operand order checked by reading, not run on hardware
//
#include <hip/hip_runtime.h>
#include <hip/hip_bf16.h>

typedef __attribute__((ext_vector_type(16))) __bf16 v16bf;
typedef __attribute__((ext_vector_type(8)))  __bf16 v8bf;
typedef __attribute__((ext_vector_type(8)))  float  v8f;

// ---------------- problem constants ----------------
constexpr long long Bn = 262144;                        // batch
constexpr int DIN[5] = {208,104,52,26,10};
constexpr int DSV[5] = {104,52,26,10,3};
constexpr int DT[5]  = {312,156,78,36,13};              // D+S combined K
constexpr int NTv[5] = {26,13,7,3,1};                   // ceil(4*S/16)  N tiles
constexpr int KCv[5] = {10,5,3,2,1};                    // ceil((D+S)/32) K chunks
constexpr int FXv[5] = {6,3,1,0,0};                     // floor(D/32) full-x chunks
constexpr int SXP[5] = {208,104,56,32,16};              // x row stride in LDS (8-mult)
// packed bf16 fragment offsets (elements); each fragment = 512 bf16 = 1KB
constexpr int WO[5] = {0, 133120, 166400, 177152, 180224};
constexpr int FRAG_ELEMS_TOTAL = 180736;                // 353 fragments
constexpr int BIAS_BYTE_OFF = FRAG_ELEMS_TOTAL * 2;     // 361472
constexpr int BO[5] = {0, 416, 624, 736, 784};          // padded bias offsets (floats)
// output offsets (elements) : c1..c5 then h1..h5
constexpr long long CO[5] = {0, Bn*104, Bn*156, Bn*182, Bn*192};
constexpr long long HO[5] = {Bn*195, Bn*299, Bn*351, Bn*377, Bn*387};

// LDS per wave (bf16): xbuf 32x208 + pre 32x416 -> 19968 elems = 39936 B
constexpr int XB_EL   = 32 * 208;                 // 6656
constexpr int PRE_EL  = 32 * 416;                 // 13312
constexpr int WAVE_EL = XB_EL + PRE_EL;           // 19968
constexpr int SMEM_BYTES = WAVE_EL * 2 * 8;       // 319488 B < 320 KB WGP LDS

// A/B fragment K pattern per ISA 16-bit layout:
// lanes 0-15: K = t (t<8) else t+8 ; lanes 16-31: +8
__device__ __forceinline__ int kpat(int t, int lane) {
    return t + ((t >> 3) << 3) + ((lane >> 4) << 3);
}

// ---------------- pre-pack kernels ----------------
// pack combined [Wx | Wh] (row n, col kk over D+S) into WMMA B fragments
__global__ void pack_frags_comb(const float* __restrict__ Wx, const float* __restrict__ Wh,
                                __bf16* __restrict__ dst,
                                int rows, int din, int ds, int nKc, int nElems) {
    int e = blockIdx.x * 256 + threadIdx.x;
    if (e >= nElems) return;
    int f    = e >> 9;
    int r    = e & 511;
    int lane = r >> 4;
    int t    = r & 15;
    int nt   = f / nKc;
    int kc   = f - nt * nKc;
    int n    = nt * 16 + (lane & 15);
    int kk   = kc * 32 + kpat(t, lane);
    float v = 0.f;
    if (n < rows) {
        if (kk < din)            v = Wx[(long long)n * din + kk];
        else if (kk < din + ds)  v = Wh[(long long)n * ds + (kk - din)];
    }
    dst[e] = (__bf16)v;
}

__global__ void pack_bias(const float* __restrict__ bx, const float* __restrict__ bh,
                          float* __restrict__ dst, int n4, int pad) {
    int i = blockIdx.x * 256 + threadIdx.x;
    if (i >= pad) return;
    dst[i] = (i < n4) ? (bx[i] + bh[i]) : 0.f;
}

// ---------------- main LSTM kernel ----------------
struct LstmArgs {
    const float* x0;
    const float* c[5];
    const float* h[5];
    const __bf16* frags;
    const float* bias;
    float* out;
};

__device__ __forceinline__ float fsig(float x)  { return 1.f / (1.f + __expf(-x)); }
__device__ __forceinline__ float ftanh(float x) { return 2.f / (1.f + __expf(-2.f * x)) - 1.f; }

union AFrag { v16bf v; v8bf h[2]; __bf16 e[16]; };

__device__ __forceinline__ v16bf ldfrag(const __bf16* base, int fragIdx, int lane) {
    return ((const v16bf*)(base + ((long long)fragIdx << 9)))[lane];
}

__global__ void __launch_bounds__(256)
lstm_stack_kernel(LstmArgs a) {
    extern __shared__ char smemraw[];
    const int lane = threadIdx.x & 31;
    const int wave = threadIdx.x >> 5;
    __bf16* xbuf = (__bf16*)smemraw + wave * WAVE_EL;   // bf16 x rows
    __bf16* pre  = xbuf + XB_EL;                        // bf16 gate pre-activations

    const long long m0 = ((long long)blockIdx.x * 8 + wave) * 32;  // 32 batch rows/wave
    const int mrow  = lane & 15;        // A / C-matrix lane row id
    const int khalf = (lane >> 4) << 3; // K half offset for A/B layout

    // stage cell-0 input rows into LDS as bf16
    for (int m = 0; m < 32; ++m)
        for (int k = lane; k < 208; k += 32)
            xbuf[m * 208 + k] = (__bf16)a.x0[(m0 + m) * 208 + k];

    AFrag ax[2][10];
    v16bf bq[2][10];

#pragma unroll
    for (int ci = 0; ci < 5; ++ci) {
        const int din = DIN[ci];
        const int ds  = DSV[ci];
        const int dt  = DT[ci];
        const int NT  = NTv[ci];
        const int KC  = KCv[ci];
        const int FX  = FXv[ci];
        const int SX  = SXP[ci];
        const int N4  = NT * 16;

        // ---- A fragments over combined K = [x | h] ----
        {
            const float* hin = a.h[ci];
#pragma unroll
            for (int mt = 0; mt < 2; ++mt) {
                const int row = mt * 16 + mrow;
                const long long hb = (m0 + row) * ds;
#pragma unroll
                for (int kc = 0; kc < KC; ++kc) {
                    if (kc < FX) {  // fully inside x: two aligned b128 LDS loads
                        int be = row * SX + kc * 32 + khalf;
                        ax[mt][kc].h[0] = *(const v8bf*)(xbuf + be);
                        ax[mt][kc].h[1] = *(const v8bf*)(xbuf + be + 16);
                    } else {        // boundary / h part: clamped gather + select
#pragma unroll
                        for (int t = 0; t < 16; ++t) {
                            int kk = kc * 32 + t + ((t >> 3) << 3) + khalf;
                            int kx = (kk < din) ? kk : 0;
                            int kh = kk - din;
                            kh = (kh < 0) ? 0 : ((kh >= ds) ? ds - 1 : kh);
                            __bf16 xv = xbuf[row * SX + kx];
                            __bf16 hv = (__bf16)hin[hb + kh];
                            ax[mt][kc].e[t] =
                                (kk < din) ? xv : ((kk < dt) ? hv : (__bf16)0.f);
                        }
                    }
                }
            }
        }

        const __bf16* wf = a.frags + WO[ci];

        // ---- prime ping-pong pipeline: fragments for nt = 0 ----
#pragma unroll
        for (int i = 0; i < KC; ++i) bq[0][i] = ldfrag(wf, i, lane);

#pragma unroll
        for (int nt = 0; nt < NT; ++nt) {
            const int cb = nt & 1;     // constant after full unroll (no copies)
            if (nt + 1 < NT) {
#pragma unroll
                for (int i = 0; i < KC; ++i)
                    bq[cb ^ 1][i] = ldfrag(wf, (nt + 1) * KC + i, lane);
            }

            v8f acc0 = {};             // SRC2 folds to inline 0 for first WMMA
            v8f acc1 = {};
#pragma unroll
            for (int i = 0; i < KC; ++i) {
                acc0 = __builtin_amdgcn_wmma_f32_16x16x32_bf16(
                    false, ax[0][i].v, false, bq[cb][i], (short)0, acc0, false, false);
                acc1 = __builtin_amdgcn_wmma_f32_16x16x32_bf16(
                    false, ax[1][i].v, false, bq[cb][i], (short)0, acc1, false, false);
            }

            // spill C/D fragments to bf16 pre (VGPR r: M = r + 8*(lane>=16), N = lane&15)
#pragma unroll
            for (int r = 0; r < 8; ++r) {
                pre[(r + khalf) * N4 + nt * 16 + mrow]      = (__bf16)acc0[r];
                pre[(16 + r + khalf) * N4 + nt * 16 + mrow] = (__bf16)acc1[r];
            }
        }

        // ---- elementwise LSTM epilogue (wave-private LDS, no barrier) ----
        // j-outer so the 4 bias loads hoist out of the m loop
        const float* cin  = a.c[ci];
        const float* bs   = a.bias + BO[ci];
        float* cout = a.out + CO[ci];
        float* hout = a.out + HO[ci];
        const int SXN = (ci < 4) ? SXP[ci + 1] : 0;   // next cell's x stride
        for (int j = lane; j < ds; j += 32) {
            const float bf_ = bs[j];
            const float bi_ = bs[ds + j];
            const float bg_ = bs[2 * ds + j];
            const float bo_ = bs[3 * ds + j];
            for (int m = 0; m < 32; ++m) {
                const int rowb = m * N4;
                const long long gb = (m0 + m) * ds;
                float fg = (float)pre[rowb + j]          + bf_;
                float ig = (float)pre[rowb + ds + j]     + bi_;
                float gg = (float)pre[rowb + 2 * ds + j] + bg_;
                float og = (float)pre[rowb + 3 * ds + j] + bo_;
                float c  = cin[gb + j];
                float cn = c * fsig(fg) + fsig(ig) * ftanh(gg);
                float hn = ftanh(cn) * fsig(og);
                cout[gb + j] = cn;
                hout[gb + j] = hn;
                if (ci < 4) xbuf[m * SXN + j] = (__bf16)hn;   // next cell's x
            }
        }
    }
}

// ---------------- launcher ----------------
extern "C" void kernel_launch(void* const* d_in, const int* in_sizes, int n_in,
                              void* d_out, int out_size, void* d_ws, size_t ws_size,
                              hipStream_t stream) {
    (void)in_sizes; (void)n_in; (void)out_size; (void)ws_size;
    // d_in order: 0 input | 1..5 c | 6..10 h | 11..15 Wx | 16..20 Wh | 21..25 bx | 26..30 bh
    __bf16* fr   = (__bf16*)d_ws;
    float*  bias = (float*)((char*)d_ws + BIAS_BYTE_OFF);

    for (int i = 0; i < 5; ++i) {
        const int rows = 4 * DSV[i];
        {   // combined [Wx | Wh] fragments
            int ne = NTv[i] * KCv[i] * 512;
            pack_frags_comb<<<(ne + 255) / 256, 256, 0, stream>>>(
                (const float*)d_in[11 + i], (const float*)d_in[16 + i],
                fr + WO[i], rows, DIN[i], DSV[i], KCv[i], ne);
        }
        {   // combined bias
            int pad = NTv[i] * 16;
            pack_bias<<<(pad + 255) / 256, 256, 0, stream>>>(
                (const float*)d_in[21 + i], (const float*)d_in[26 + i],
                bias + BO[i], rows, pad);
        }
    }

    LstmArgs args;
    args.x0 = (const float*)d_in[0];
    for (int i = 0; i < 5; ++i) {
        args.c[i] = (const float*)d_in[1 + i];
        args.h[i] = (const float*)d_in[6 + i];
    }
    args.frags = fr;
    args.bias  = bias;
    args.out   = (float*)d_out;

    // 1024 blocks * 8 waves * 32 rows = 262144
    lstm_stack_kernel<<<1024, 256, SMEM_BYTES, stream>>>(args);
}
